// Batch_MGCN_Value_8495445311632
// MI455X (gfx1250) — matrix operand order repaired and checked by reading, not compiled, not run on hardware
//
#include <hip/hip_runtime.h>

// ---------------------------------------------------------------------------
// CDNA5 (gfx1250) WMMA types / helpers
// ---------------------------------------------------------------------------
typedef __attribute__((ext_vector_type(16))) __bf16        v16bf;
typedef __attribute__((ext_vector_type(8)))  float         v8f;
typedef __attribute__((ext_vector_type(4)))  unsigned int  uiv4;
typedef __attribute__((ext_vector_type(4)))  float         fv4;

#define LRELU_SLOPE 0.01f

__device__ __forceinline__ __bf16 f2bf(float x) { return (__bf16)x; }

__device__ __forceinline__ unsigned pk2(float a, float b) {
    union { __bf16 h[2]; unsigned u; } p;
    p.h[0] = (__bf16)a; p.h[1] = (__bf16)b;
    return p.u;   // lowers to v_cvt_pk_bf16_f32
}
// pack 8 consecutive floats -> 8 bf16 (one 16B LDS store)
__device__ __forceinline__ uiv4 pack8(fv4 a, fv4 b) {
    uiv4 v;
    v[0] = pk2(a[0], a[1]); v[1] = pk2(a[2], a[3]);
    v[2] = pk2(b[0], b[1]); v[3] = pk2(b[2], b[3]);
    return v;
}

// A fragment: 16x32 bf16 tile, row-major in LDS (ISA 7.12.2 layout).
__device__ __forceinline__ v16bf load_frag_a(const __bf16* p) {
    union { v16bf v; uiv4 u[2]; } f;
    f.u[0] = *(const uiv4*)(p);
    f.u[1] = *(const uiv4*)(p + 16);
    return f.v;
}
// B fragment: 32x16 tile stored K-contiguous per output column (WT[n][k]).
__device__ __forceinline__ v16bf load_frag_b(const __bf16* p) {
    union { v16bf v; uiv4 u[2]; } f;
    f.u[0] = *(const uiv4*)(p);
    f.u[1] = *(const uiv4*)(p + 16);
    return f.v;
}

// ---------------------------------------------------------------------------
// One layer: Out[m][n] = lrelu(A(64xK) @ W(KxN) + bias), 8 waves, 64 rows.
// ---------------------------------------------------------------------------
template <int K, int NCOLS>
__device__ __forceinline__ void layer_lds(const __bf16* A, int lda,
                                          const __bf16* WT, int ldw,
                                          const float* bias,
                                          __bf16* Out, int ldo,
                                          int wave, int lane) {
    const int ntiles = 4 * (NCOLS / 16);
    for (int t = wave; t < ntiles; t += 8) {
        const int m0 = (t & 3) << 4;
        const int n0 = (t >> 2) << 4;
        v8f c = {};
#pragma unroll
        for (int k0 = 0; k0 < K; k0 += 32) {
            v16bf a = load_frag_a(A + (m0 + (lane & 15)) * lda + k0 + ((lane >> 4) << 3));
            v16bf b = load_frag_b(WT + (n0 + (lane & 15)) * ldw + k0 + ((lane >> 4) << 3));
            c = __builtin_amdgcn_wmma_f32_16x16x32_bf16(false, a, false, b,
                                                        (short)0, c, false, false);
        }
        const int col = n0 + (lane & 15);
        const int rb  = m0 + ((lane >> 4) << 3);
        const float bv = bias[col];
#pragma unroll
        for (int r = 0; r < 8; ++r) {
            float v = c[r] + bv;
            v = (v >= 0.f) ? v : LRELU_SLOPE * v;
            Out[(rb + r) * ldo + col] = f2bf(v);
        }
    }
}

// Final layer -> global f32 (optionally accumulate). LeakyReLU always on
// (all WMMA-path MLPs here have final_act=True; 'g' head is separate).
template <int K, int NCOLS>
__device__ __forceinline__ void layer_global(const __bf16* A, int lda,
                                             const __bf16* WT, int ldw,
                                             const float* bias,
                                             float* __restrict__ Y, int ldy,
                                             long rowBase, int wave, int lane,
                                             int accum) {
    const int ntiles = 4 * (NCOLS / 16);
    for (int t = wave; t < ntiles; t += 8) {
        const int m0 = (t & 3) << 4;
        const int n0 = (t >> 2) << 4;
        v8f c = {};
#pragma unroll
        for (int k0 = 0; k0 < K; k0 += 32) {
            v16bf a = load_frag_a(A + (m0 + (lane & 15)) * lda + k0 + ((lane >> 4) << 3));
            v16bf b = load_frag_b(WT + (n0 + (lane & 15)) * ldw + k0 + ((lane >> 4) << 3));
            c = __builtin_amdgcn_wmma_f32_16x16x32_bf16(false, a, false, b,
                                                        (short)0, c, false, false);
        }
        const int col = n0 + (lane & 15);
        const int rb  = m0 + ((lane >> 4) << 3);
        const float bv = bias[col];
#pragma unroll
        for (int r = 0; r < 8; ++r) {
            float v = c[r] + bv;
            v = (v >= 0.f) ? v : LRELU_SLOPE * v;
            float* p = Y + (rowBase + rb + r) * (long)ldy + col;
            if (accum) *p += v; else *p = v;
        }
    }
}

// ---------------------------------------------------------------------------
// One-shot weight prep: f32 [K][N] -> bf16 transposed [N][LD] (K zero-padded),
// exactly the LDS layout, so blocks stage weights as raw 16B copies.
// ---------------------------------------------------------------------------
struct PrepEnt { const float* W; int dstOff; int K; int LD; int N; };
struct PrepTable { PrepEnt e[21]; };

__global__ __launch_bounds__(256)
void prep_weights_kernel(PrepTable tab, __bf16* __restrict__ dst) {
    PrepEnt e = tab.e[blockIdx.x];
    const int total = e.N * e.LD;
    for (int i = threadIdx.x; i < total; i += 256) {
        int n = i / e.LD, k = i % e.LD;
        dst[e.dstOff + i] = (k < e.K) ? f2bf(e.W[k * e.N + n]) : f2bf(0.f);
    }
}

// ---------------------------------------------------------------------------
// Fused 3-layer MLP, 64 rows per 256-thread block, prepped bf16 weights.
// ---------------------------------------------------------------------------
template <int D0, int D1, int D2, int D3>
__global__ __launch_bounds__(256)
void mlp3_kernel(const float* __restrict__ X, long xBatchStride, int rowsPerBatch,
                 const __bf16* __restrict__ WT0, const float* __restrict__ B0,
                 const __bf16* __restrict__ WT1, const float* __restrict__ B1,
                 const __bf16* __restrict__ WT2, const float* __restrict__ B2,
                 float* __restrict__ Y, int accumOut) {
    constexpr int K0  = (D0 < 32) ? 32 : D0;
    constexpr int LD0 = K0 + 8;
    constexpr int LD1 = D1 + 8;
    constexpr int LD2 = D2 + 8;

    __shared__ __align__(16) __bf16 sX [64][LD0];
    __shared__ __align__(16) __bf16 sW0[D1][LD0];
    __shared__ __align__(16) __bf16 sW1[D2][LD1];
    __shared__ __align__(16) __bf16 sW2[D3][LD2];
    __shared__ __align__(16) __bf16 sA1[64][LD1];
    __shared__ __align__(16) __bf16 sA2[64][LD2];
    __shared__ float sB0[D1], sB1[D2], sB2[D3];

    const int tid  = threadIdx.x;
    const int wave = tid >> 5;
    const int lane = tid & 31;
    const long rowBase = (long)blockIdx.x * 64;

    // weights: raw vector copies of the prepped layout
    for (int i = tid; i < (D1 * LD0) / 8; i += 256)
        ((uiv4*)&sW0[0][0])[i] = ((const uiv4*)WT0)[i];
    for (int i = tid; i < (D2 * LD1) / 8; i += 256)
        ((uiv4*)&sW1[0][0])[i] = ((const uiv4*)WT1)[i];
    for (int i = tid; i < (D3 * LD2) / 8; i += 256)
        ((uiv4*)&sW2[0][0])[i] = ((const uiv4*)WT2)[i];
    for (int i = tid; i < D1; i += 256) sB0[i] = B0[i];
    for (int i = tid; i < D2; i += 256) sB1[i] = B1[i];
    for (int i = tid; i < D3; i += 256) sB2[i] = B2[i];

    // input rows: 8 floats per unit -> one 16B bf16 store
    constexpr int UPR = K0 / 8;
    for (int i = tid; i < 64 * UPR; i += 256) {
        int lr = i / UPR, k8 = (i % UPR) * 8;
        long r = rowBase + lr;
        long bb = r / rowsPerBatch;
        long nn = r % rowsPerBatch;
        uiv4 v = (uiv4)0;
        if (k8 < D0) {
            const float* xp = X + bb * xBatchStride + nn * (long)D0 + k8;
            v = pack8(*(const fv4*)xp, *(const fv4*)(xp + 4));
        }
        *(uiv4*)&sX[lr][k8] = v;
    }
    __syncthreads();

    layer_lds<K0, D1>(&sX[0][0],  LD0, &sW0[0][0], LD0, sB0, &sA1[0][0], LD1, wave, lane);
    __syncthreads();
    layer_lds<D1, D2>(&sA1[0][0], LD1, &sW1[0][0], LD1, sB1, &sA2[0][0], LD2, wave, lane);
    __syncthreads();
    layer_global<D2, D3>(&sA2[0][0], LD2, &sW2[0][0], LD2, sB2, Y, D3,
                         rowBase, wave, lane, accumOut);
}

// ---------------------------------------------------------------------------
// Both message MLPs fused: stage h tile once, run chain A -> M0, chain B -> M1.
// ---------------------------------------------------------------------------
__global__ __launch_bounds__(256)
void msg_dual_kernel(const float* __restrict__ Xg,
                     const __bf16* Wa0, const float* Ba0,
                     const __bf16* Wa1, const float* Ba1,
                     const __bf16* Wa2, const float* Ba2,
                     const __bf16* Wb0, const float* Bb0,
                     const __bf16* Wb1, const float* Bb1,
                     const __bf16* Wb2, const float* Bb2,
                     float* __restrict__ M0, float* __restrict__ M1) {
    constexpr int LD = 72;
    __shared__ __align__(16) __bf16 sX [64][LD];
    __shared__ __align__(16) __bf16 sW [6][64 * LD];
    __shared__ __align__(16) __bf16 sA1[64][LD];
    __shared__ __align__(16) __bf16 sA2[64][LD];
    __shared__ float sB[6][64];

    const int tid  = threadIdx.x;
    const int wave = tid >> 5;
    const int lane = tid & 31;
    const long rowBase = (long)blockIdx.x * 64;

    const __bf16* Ws[6] = { Wa0, Wa1, Wa2, Wb0, Wb1, Wb2 };
    const float*  Bs[6] = { Ba0, Ba1, Ba2, Bb0, Bb1, Bb2 };
#pragma unroll
    for (int w = 0; w < 6; ++w) {
        for (int i = tid; i < (64 * LD) / 8; i += 256)
            ((uiv4*)&sW[w][0])[i] = ((const uiv4*)Ws[w])[i];
        if (tid < 64) sB[w][tid] = Bs[w][tid];
    }
    for (int i = tid; i < 64 * 8; i += 256) {          // 64 rows x 8 units
        int lr = i >> 3, k8 = (i & 7) << 3;
        const float* xp = Xg + (rowBase + lr) * 64 + k8;
        *(uiv4*)&sX[lr][k8] = pack8(*(const fv4*)xp, *(const fv4*)(xp + 4));
    }
    __syncthreads();

    // chain A
    layer_lds<64, 64>(&sX[0][0], LD, &sW[0][0], LD, sB[0], &sA1[0][0], LD, wave, lane);
    __syncthreads();
    layer_lds<64, 64>(&sA1[0][0], LD, &sW[1][0], LD, sB[1], &sA2[0][0], LD, wave, lane);
    __syncthreads();
    layer_global<64, 64>(&sA2[0][0], LD, &sW[2][0], LD, sB[2], M0, 64, rowBase, wave, lane, 0);
    // chain B (sA1 free: its readers finished before previous barrier)
    layer_lds<64, 64>(&sX[0][0], LD, &sW[3][0], LD, sB[3], &sA1[0][0], LD, wave, lane);
    __syncthreads();   // all waves done with A-L3 reads of sA2 + B-L1 writes
    layer_lds<64, 64>(&sA1[0][0], LD, &sW[4][0], LD, sB[4], &sA2[0][0], LD, wave, lane);
    __syncthreads();
    layer_global<64, 64>(&sA2[0][0], LD, &sW[5][0], LD, sB[5], M1, 64, rowBase, wave, lane, 0);
}

// ---------------------------------------------------------------------------
// Fused bmm + aggregation MLP + residual:
//   h[b, n0:n0+64] += mlp_agg( sum_t adj[b,t] @ M_t[b] )  for this row block.
// grid = (N/64, B), 256 threads (8 waves), each wave owns two 16x16 tiles.
// ---------------------------------------------------------------------------
__global__ __launch_bounds__(256)
void bmm_agg_kernel(const float* __restrict__ adj,   // [B,T,N,N]
                    const float* __restrict__ M0g,   // [B,N,H]
                    const float* __restrict__ M1g,
                    const __bf16* W0, const float* B0,
                    const __bf16* W1, const float* B1,
                    const __bf16* W2, const float* B2,
                    float* __restrict__ h) {         // [B,N,H] (+=)
    constexpr int N = 1024, H = 64, T = 2;
    constexpr int LDA = 40, LDB = 40, LD = 72;
    __shared__ __align__(16) __bf16 sAdj[64][LDA];
    __shared__ __align__(16) __bf16 sMt [64][LDB];   // transposed: [col][k]
    __shared__ __align__(16) __bf16 sW0a[64][LD], sW1a[64][LD], sW2a[64][LD];
    __shared__ __align__(16) __bf16 sM  [64][LD];    // msgs tile, reused as A2
    __shared__ __align__(16) __bf16 sA1 [64][LD];
    __shared__ float sB0[64], sB1[64], sB2[64];

    const int tid  = threadIdx.x;
    const int wave = tid >> 5;
    const int lane = tid & 31;
    const int n0   = blockIdx.x * 64;
    const int b    = blockIdx.y;

    for (int i = tid; i < (64 * LD) / 8; i += 256) {
        ((uiv4*)&sW0a[0][0])[i] = ((const uiv4*)W0)[i];
        ((uiv4*)&sW1a[0][0])[i] = ((const uiv4*)W1)[i];
        ((uiv4*)&sW2a[0][0])[i] = ((const uiv4*)W2)[i];
    }
    if (tid < 64) { sB0[tid] = B0[tid]; sB1[tid] = B1[tid]; sB2[tid] = B2[tid]; }

    const float* Ms[2]  = { M0g + (size_t)b * N * H, M1g + (size_t)b * N * H };
    const float* adjB   = adj + (size_t)b * T * N * N;

    v8f acc0 = {}, acc1 = {};
    for (int k0 = 0; k0 < N; k0 += 32) {
        for (int t = 0; t < T; ++t) {
            __syncthreads();   // previous chunk's fragment reads done
            {   // adjacency tile 64x32: exactly one 8-float unit per thread
                int r = tid >> 2, k8 = (tid & 3) << 3;
                const float* src = adjB + ((size_t)t * N + (n0 + r)) * N + k0 + k8;
                *(uiv4*)&sAdj[r][k8] = pack8(*(const fv4*)src, *(const fv4*)(src + 4));
            }
            {   // M_t tile 32x64 transposed: one (2k x 4c) unit per thread
                int k2 = (tid >> 4) << 1, c4 = (tid & 15) << 2;
                const float* m = Ms[t] + (size_t)(k0 + k2) * H + c4;
                fv4 r0 = *(const fv4*)m;
                fv4 r1 = *(const fv4*)(m + H);
#pragma unroll
                for (int j = 0; j < 4; ++j)
                    *(unsigned*)&sMt[c4 + j][k2] = pk2(r0[j], r1[j]);
            }
            if (k0 + 32 < N)   // pull next adj chunk toward L2
                __builtin_prefetch(adjB + ((size_t)t * N + (n0 + (tid >> 2))) * N + k0 + 32, 0, 1);
            __syncthreads();
            {
                const int t0 = wave, t1 = wave + 8;
                v16bf a = load_frag_a(&sAdj[0][0] + (((t0 & 3) << 4) + (lane & 15)) * LDA + ((lane >> 4) << 3));
                v16bf bb = load_frag_b(&sMt[0][0] + (((t0 >> 2) << 4) + (lane & 15)) * LDB + ((lane >> 4) << 3));
                acc0 = __builtin_amdgcn_wmma_f32_16x16x32_bf16(false, a, false, bb, (short)0, acc0, false, false);
                a  = load_frag_a(&sAdj[0][0] + (((t1 & 3) << 4) + (lane & 15)) * LDA + ((lane >> 4) << 3));
                bb = load_frag_b(&sMt[0][0] + (((t1 >> 2) << 4) + (lane & 15)) * LDB + ((lane >> 4) << 3));
                acc1 = __builtin_amdgcn_wmma_f32_16x16x32_bf16(false, a, false, bb, (short)0, acc1, false, false);
            }
        }
    }

    // msgs tile -> LDS bf16 (raw, no bias/act), then the 3-layer agg MLP
    {
        int tt = wave;
        int col = ((tt >> 2) << 4) + (lane & 15);
        int rb  = ((tt & 3) << 4) + ((lane >> 4) << 3);
#pragma unroll
        for (int r = 0; r < 8; ++r) sM[rb + r][col] = f2bf(acc0[r]);
        tt = wave + 8;
        col = ((tt >> 2) << 4) + (lane & 15);
        rb  = ((tt & 3) << 4) + ((lane >> 4) << 3);
#pragma unroll
        for (int r = 0; r < 8; ++r) sM[rb + r][col] = f2bf(acc1[r]);
    }
    __syncthreads();
    layer_lds<64, 64>(&sM[0][0], LD, &sW0a[0][0], LD, sB0, &sA1[0][0], LD, wave, lane);
    __syncthreads();
    layer_lds<64, 64>(&sA1[0][0], LD, &sW1a[0][0], LD, sB1, &sM[0][0], LD, wave, lane);
    __syncthreads();
    layer_global<64, 64>(&sM[0][0], LD, &sW2a[0][0], LD, sB2, h, H,
                         (long)b * N + n0, wave, lane, /*accum=*/1);
}

// ---------------------------------------------------------------------------
// gather / pool / tiny g head
// ---------------------------------------------------------------------------
__global__ void gather_kernel(const float* __restrict__ gcn,
                              const int* __restrict__ idx,
                              float* __restrict__ sw, int total) {
    int i = blockIdx.x * blockDim.x + threadIdx.x;
    if (i >= total) return;
    int o  = i & 31;
    int bs = i >> 5;
    int b  = bs >> 8;
    int node = idx[i];
    sw[i] = gcn[((size_t)b * 1024 + node) * 32 + o];
}

__global__ void pool_kernel(const float* __restrict__ fs, float* __restrict__ pooled) {
    int i = blockIdx.x * blockDim.x + threadIdx.x;
    if (i >= 16 * 32) return;
    int b = i >> 5, o = i & 31;
    float s = 0.f;
    for (int ss = 0; ss < 256; ++ss) s += fs[((size_t)b * 256 + ss) * 32 + o];
    pooled[i] = s;
}

__global__ __launch_bounds__(64)
void mlp_g_kernel(const float* __restrict__ pooled,
                  const float* __restrict__ W0, const float* __restrict__ B0,
                  const float* __restrict__ W1, const float* __restrict__ B1,
                  const float* __restrict__ W2, const float* __restrict__ B2,
                  float* __restrict__ out) {
    int b = blockIdx.x, j = threadIdx.x;
    __shared__ float x[32], h1[64], h2[64];
    if (j < 32) x[j] = pooled[b * 32 + j];
    __syncthreads();
    float acc = B0[j];
    for (int k = 0; k < 32; ++k) acc += x[k] * W0[k * 64 + j];
    h1[j] = (acc >= 0.f) ? acc : LRELU_SLOPE * acc;
    __syncthreads();
    acc = B1[j];
    for (int k = 0; k < 64; ++k) acc += h1[k] * W1[k * 64 + j];
    h2[j] = (acc >= 0.f) ? acc : LRELU_SLOPE * acc;
    __syncthreads();
    if (j == 0) {
        float o = B2[0];
        for (int k = 0; k < 64; ++k) o += h2[k] * W2[k];
        out[b] = o;   // final_act = False
    }
}

// ---------------------------------------------------------------------------
// Orchestration
// ---------------------------------------------------------------------------
extern "C" void kernel_launch(void* const* d_in, const int* in_sizes, int n_in,
                              void* d_out, int out_size, void* d_ws, size_t ws_size,
                              hipStream_t stream) {
    constexpr int B = 16, T = 2, N = 1024, F = 16, H = 64, OUT = 32, S = 256;
    constexpr long R = (long)B * N;
    if (n_in < 51) return;

    const float* node_feats = (const float*)d_in[0];
    const float* adj        = (const float*)d_in[1];
    const int*   sidx       = (const int*)d_in[2];
    auto P = [&](int i) { return (const float*)d_in[3 + i]; };
    // leaf order (dict insertion): in0(0-5) in1(6-11) msg0(12-17) msg1(18-23)
    // agg(24-29) out(30-35) f(36-41) g(42-47); each W0,b0,W1,b1,W2,b2

    float* h      = (float*)d_ws;
    float* M0     = h    + R * H;
    float* M1     = M0   + R * H;
    float* gcn    = M1   + R * H;
    float* sw     = gcn  + R * OUT;
    float* fs     = sw   + (long)B * S * OUT;
    float* pooled = fs   + (long)B * S * OUT;
    __bf16* wbf   = (__bf16*)(pooled + (long)B * OUT);   // 16B-aligned (even f32 count)

    // --- build weight-prep table (7 WMMA MLPs x 3 layers) ---
    const int baseIdx[7] = { 0, 6, 12, 18, 24, 30, 36 };  // in0,in1,msg0,msg1,agg,out,f
    const int K0s[7]     = { 16, 16, 64, 64, 64, 64, 32 };
    const int D3s[7]     = { 64, 64, 64, 64, 64, 32, 32 };
    PrepTable tab;
    int offs[7][3];
    int cur = 0;
    for (int m = 0; m < 7; ++m) {
        for (int l = 0; l < 3; ++l) {
            int K  = (l == 0) ? K0s[m] : 64;
            int Nn = (l == 2) ? D3s[m] : 64;
            int Kp = (K < 32) ? 32 : K;
            int LD = Kp + 8;
            tab.e[m * 3 + l] = { P(baseIdx[m] + 2 * l), cur, K, LD, Nn };
            offs[m][l] = cur;
            cur += Nn * LD;
        }
    }
    auto WB = [&](int m, int l) { return (const __bf16*)(wbf + offs[m][l]); };

    prep_weights_kernel<<<21, 256, 0, stream>>>(tab, wbf);

    // 1) h = sum_t mlp_in[t](node_feats[:, t])
    mlp3_kernel<F, H, H, H><<<R / 64, 256, 0, stream>>>(
        node_feats, (long)T * N * F, N,
        WB(0, 0), P(1), WB(0, 1), P(3), WB(0, 2), P(5), h, 0);
    mlp3_kernel<F, H, H, H><<<R / 64, 256, 0, stream>>>(
        node_feats + (long)N * F, (long)T * N * F, N,
        WB(1, 0), P(7), WB(1, 1), P(9), WB(1, 2), P(11), h, 1);

    // 2) message passing: msg MLPs fused; bmm fused with agg MLP + residual
    for (int step = 0; step < 2; ++step) {
        msg_dual_kernel<<<R / 64, 256, 0, stream>>>(
            h,
            WB(2, 0), P(13), WB(2, 1), P(15), WB(2, 2), P(17),
            WB(3, 0), P(19), WB(3, 1), P(21), WB(3, 2), P(23),
            M0, M1);
        bmm_agg_kernel<<<dim3(N / 64, B), 256, 0, stream>>>(
            adj, M0, M1,
            WB(4, 0), P(25), WB(4, 1), P(27), WB(4, 2), P(29), h);
    }

    // 3) gcn_out = mlp_out(h)
    mlp3_kernel<H, H, H, OUT><<<R / 64, 256, 0, stream>>>(
        h, 0, (int)R, WB(5, 0), P(31), WB(5, 1), P(33), WB(5, 2), P(35), gcn, 0);

    // 4) gather along N
    gather_kernel<<<(B * S * OUT) / 256, 256, 0, stream>>>(gcn, sidx, sw, B * S * OUT);

    // 5) pooled = mlp_f(switch).sum(axis=1)
    mlp3_kernel<OUT, H, H, OUT><<<(B * S) / 64, 256, 0, stream>>>(
        sw, 0, B * S, WB(6, 0), P(37), WB(6, 1), P(39), WB(6, 2), P(41), fs, 0);
    pool_kernel<<<2, 256, 0, stream>>>(fs, pooled);

    // 6) out = mlp_g(pooled), no final act
    mlp_g_kernel<<<B, 64, 0, stream>>>(pooled, P(42), P(43), P(44), P(45), P(46), P(47),
                                       (float*)d_out);
}
